// SAVSS_2D_DirConv_45681272160342
// MI455X (gfx1250) — compile-verified, matrix operand order checked by reading, not run on hardware
//
#include <hip/hip_runtime.h>
#include <hip/hip_bf16.h>

// ---------------- problem constants (fixed by reference) ----------------
#define DIRS   4
#define BB     2
#define HH     48
#define WW     48
#define LL     (HH*WW)        // 2304
#define DM     128
#define DI     256            // d_inner
#define NS     16             // d_state
#define DTR    8              // dt_rank
#define XDBL_LD 48            // padded x_dbl leading dim (40 -> 48)
#define NCH    32             // scan chunks
#define CHUNK  (LL/NCH)       // 72
#define ML     (BB*LL)        // 4608 rows in flattened GEMMs

#if __has_builtin(__builtin_amdgcn_tensor_load_to_lds)
#define HAVE_TDM 1
#else
#define HAVE_TDM 0
#endif
#if __has_include(<hip/amd_detail/amd_gfx1250_TDM.h>)
#define TDM_ARGS6 1
#else
#define TDM_ARGS6 0
#endif

typedef __attribute__((ext_vector_type(16))) __bf16       v16bf;
typedef __attribute__((ext_vector_type(8)))  float        v8f;
typedef __attribute__((ext_vector_type(4)))  unsigned int v4u;
typedef __attribute__((ext_vector_type(8)))  int          v8i;
typedef __attribute__((ext_vector_type(4)))  int          v4i;
typedef unsigned long long ull;

union Frag2 { v16bf v; uint4 q[2]; };

__device__ __forceinline__ unsigned short f2bf(float f) {
  unsigned int x = __float_as_uint(f);
  x += 0x7FFFu + ((x >> 16) & 1u);      // round-to-nearest-even
  return (unsigned short)(x >> 16);
}

#if HAVE_TDM
// Issue one TDM load of a 16-row x 32-col bf16 tile (row stride in elements)
// into LDS at byte offset lds_addr.  D# per CDNA5 ISA §8.3/8.4 (2D tensor,
// data_size=2B, tile==tensor extent, groups 2/3 zero).
__device__ __forceinline__ void tdm_issue(unsigned lds_addr, ull gaddr,
                                          int row_stride_elems) {
  v4u g0;
  g0.x = 1u;                                            // count=1, user desc
  g0.y = lds_addr;                                      // lds_addr [63:32]
  g0.z = (unsigned)gaddr;                               // global_addr lo
  g0.w = (unsigned)((gaddr >> 32) & 0x01FFFFFFu) | (2u << 30);  // ga hi | type=2
  v8i g1;
  g1[0] = (int)(1u << 16);          // workgroup_mask=0, data_size=1 (2 bytes)
  g1[1] = (int)(32u << 16);         // [63:48]  tensor_dim0 = 32
  g1[2] = (int)(16u << 16);         // [95:80]  tensor_dim1 = 16
  g1[3] = (int)(32u << 16);         // [127:112] tile_dim0 = 32
  g1[4] = (int)16;                  // [143:128] tile_dim1 = 16, tile_dim2 = 0
  g1[5] = (int)(unsigned)row_stride_elems;  // stride0 lo32 (data_size units)
  g1[6] = 0;                        // stride0 hi16 = 0, stride1 lo = 0
  g1[7] = 0;
  v4i z4 = {0, 0, 0, 0};
#if TDM_ARGS6
  v8i z8 = {0, 0, 0, 0, 0, 0, 0, 0};
  __builtin_amdgcn_tensor_load_to_lds(g0, g1, z4, z4, z8, 0);
#else
  __builtin_amdgcn_tensor_load_to_lds(g0, g1, z4, z4, 0);
#endif
}
#endif

// C[m,n] = sum_k A[m,k] * Bw[n,k], A/Bw bf16 row-major, C f32.
// One 16x16 A-tile x NB 16-wide N-tiles per wave; TDM double-buffered staging.
template<int NB>
__global__ __launch_bounds__(128)
void wmma_gemm_bf(const unsigned short* __restrict__ A, int lda,
                  const unsigned short* __restrict__ Bw, int ldb,
                  float* __restrict__ C, int ldc,
                  int M, int N, int K) {
#if HAVE_TDM
  __shared__ unsigned short smem[4 * 2 * (1 + NB) * 512];   // 4 waves, 2 bufs
#endif
  const int wid  = threadIdx.x >> 5;
  const int wave = blockIdx.x * (blockDim.x >> 5) + wid;
  const int tnc  = N / (16 * NB);
  const int tm = wave / tnc, tn0 = (wave % tnc) * NB;
  if (tm * 16 >= M) return;                  // wave-uniform guard
  const int lane = threadIdx.x & 31;
  const int row  = lane & 15;
  const int kb   = (lane & 16) ? 8 : 0;      // ISA 7.12.2 16-bit A/B layout
  v8f acc[NB] = {};
  const int ksteps = K / 32;

#if HAVE_TDM
  unsigned short* buf0 = smem + wid * 2 * (1 + NB) * 512;
  unsigned short* buf1 = buf0 + (1 + NB) * 512;
  auto issue = [&](int step, unsigned short* s) {
    const int k0 = step * 32;
    tdm_issue((unsigned)(uintptr_t)s,
              (ull)(uintptr_t)(A + (size_t)(tm * 16) * lda + k0), lda);
#pragma unroll
    for (int j = 0; j < NB; ++j)
      tdm_issue((unsigned)(uintptr_t)(s + (1 + j) * 512),
                (ull)(uintptr_t)(Bw + (size_t)((tn0 + j) * 16) * ldb + k0), ldb);
  };
  issue(0, buf0);
  for (int step = 0; step < ksteps; ++step) {
    unsigned short* cur = (step & 1) ? buf1 : buf0;
    unsigned short* nxt = (step & 1) ? buf0 : buf1;
    if (step + 1 < ksteps) {
      issue(step + 1, nxt);                          // overlap DMA with math
      __builtin_amdgcn_s_wait_tensorcnt(1 + NB);     // current group landed
    } else {
      __builtin_amdgcn_s_wait_tensorcnt(0);
    }
    Frag2 fa;
    fa.q[0] = *(const uint4*)(cur + row * 32 + kb);
    fa.q[1] = *(const uint4*)(cur + row * 32 + kb + 16);
#pragma unroll
    for (int j = 0; j < NB; ++j) {
      const unsigned short* sb = cur + (1 + j) * 512;
      Frag2 fb;
      fb.q[0] = *(const uint4*)(sb + row * 32 + kb);
      fb.q[1] = *(const uint4*)(sb + row * 32 + kb + 16);
      acc[j] = __builtin_amdgcn_wmma_f32_16x16x32_bf16(
          false, fa.v, false, fb.v, (short)0, acc[j], false, false);
    }
  }
#else
  const unsigned short* arow = A + (size_t)(tm * 16 + row) * lda;
  for (int step = 0; step < ksteps; ++step) {
    const int k0 = step * 32;
    Frag2 fa;
    fa.q[0] = *(const uint4*)(arow + k0 + kb);
    fa.q[1] = *(const uint4*)(arow + k0 + kb + 16);
#pragma unroll
    for (int j = 0; j < NB; ++j) {
      const unsigned short* brow = Bw + (size_t)((tn0 + j) * 16 + row) * ldb;
      Frag2 fb;
      fb.q[0] = *(const uint4*)(brow + k0 + kb);
      fb.q[1] = *(const uint4*)(brow + k0 + kb + 16);
      acc[j] = __builtin_amdgcn_wmma_f32_16x16x32_bf16(
          false, fa.v, false, fb.v, (short)0, acc[j], false, false);
    }
  }
#endif
  // C/D layout: lane = N (mod 16), reg r -> M = r + 8*(lane>=16)
  const int mbase = tm * 16 + ((lane >> 4) & 1) * 8;
#pragma unroll
  for (int j = 0; j < NB; ++j) {
    const int n = (tn0 + j) * 16 + row;
#pragma unroll
    for (int r = 0; r < 8; ++r)
      C[(size_t)(mbase + r) * ldc + n] = acc[j][r];
  }
}

// ---------------- bf16 staging converters --------------------------------
__global__ void f32_to_bf16(const float* __restrict__ in,
                            unsigned short* __restrict__ out, int n) {
  int i = blockIdx.x * blockDim.x + threadIdx.x;
  if (i < n) out[i] = f2bf(in[i]);
}
__global__ void pad_xproj_bf(const float* __restrict__ w,
                             unsigned short* __restrict__ wp) {
  int idx = blockIdx.x * blockDim.x + threadIdx.x;
  if (idx >= XDBL_LD * DI) return;
  wp[idx] = (idx < (DTR + 2 * NS) * DI) ? f2bf(w[idx]) : (unsigned short)0;
}
__global__ void extract_u_bf(const float* __restrict__ xz,
                             unsigned short* __restrict__ u) {
  int i = blockIdx.x * blockDim.x + threadIdx.x;
  if (i >= ML * DI) return;
  int m = i / DI, d = i - m * DI;
  u[i] = f2bf(xz[(size_t)m * (2 * DI) + d]);
}

// ---------------- scan-order generation (4 space-filling curves) --------
__global__ void prep_orders(int* __restrict__ ord) {
  const int tid = threadIdx.x;
  for (int t = tid; t < LL; t += blockDim.x) {
    int r = t / WW, c = t % WW;
    int j0 = (r & 1) ? (WW - 1 - c) : c;
    ord[0 * LL + t] = (HH - 1 - r) * WW + j0;       // bottom-up row snake
    int cc = t / HH, rr = t % HH;
    int i1 = (cc & 1) ? (HH - 1 - rr) : rr;
    ord[1 * LL + t] = i1 * WW + cc;                 // column snake
  }
  for (int dg = tid; dg < HH + WW - 1; dg += blockDim.x) {
    int S = (dg <= WW) ? dg * (dg + 1) / 2 : LL - (2 * WW - 1 - dg) * (2 * WW - dg) / 2;
    int i0 = max(0, dg - (WW - 1)), i1 = min(HH - 1, dg);
    int len = i1 - i0 + 1;
    for (int k = 0; k < len; ++k) {
      int i = (dg & 1) ? (i1 - k) : (i0 + k);
      ord[2 * LL + S + k] = i * WW + (dg - i);
      ord[3 * LL + S + k] = i * WW + (WW - 1 - (dg - i));
    }
  }
}

// delta = softplus(x_dbl[:, :8] @ dt_proj_w^T + dt_proj_b)
__global__ void compute_delta(const float* __restrict__ xdbl,
                              const float* __restrict__ dtw,
                              const float* __restrict__ dtb,
                              float* __restrict__ delta) {
  const int m = blockIdx.x, d = threadIdx.x;
  const float* xr = xdbl + (size_t)m * XDBL_LD;
  float acc = dtb[d];
#pragma unroll
  for (int r = 0; r < DTR; ++r) acc += xr[r] * dtw[d * DTR + r];
  delta[(size_t)m * DI + d] = (acc > 20.f) ? acc : log1pf(__expf(acc));
}

// ---------------- chunked selective scan --------------------------------
__device__ __forceinline__ void scan_decode(int wave, int lane,
                                            int& dir, int& b, int& d, int& n,
                                            int& chunk) {
  chunk = wave % NCH;  int rest = wave / NCH;
  int dpair = rest % (DI / 2); rest /= (DI / 2);
  b = rest % BB; dir = rest / BB;
  n = lane & 15;
  d = dpair * 2 + (lane >> 4);
}

__global__ void scan_pass1(const float* __restrict__ delta,
                           const float* __restrict__ xz,
                           const float* __restrict__ xdbl,
                           const float* __restrict__ A_log,
                           const float* __restrict__ dirB,
                           const int*   __restrict__ ord,
                           float* __restrict__ carryA,
                           float* __restrict__ carryB) {
  const int wave = blockIdx.x * (blockDim.x >> 5) + (threadIdx.x >> 5);
  const int lane = threadIdx.x & 31;
  int dir, b, d, n, chunk;
  scan_decode(wave, lane, dir, b, d, n, chunk);
  const float An  = -__expf(A_log[d * NS + n]);
  const float dBn = dirB[dir * NS + n];
  const int* o = ord + dir * LL;
  float aP = 1.f, h = 0.f;
  const int t0 = chunk * CHUNK;
#pragma unroll 4
  for (int t = t0; t < t0 + CHUNK; ++t) {
    int   idx = o[t];
    float dv  = delta[((size_t)(b * LL + t)) * DI + d];
    float uv  = xz[((size_t)(b * LL + idx)) * (2 * DI) + d];
    float Bn  = xdbl[((size_t)(b * LL + t)) * XDBL_LD + DTR + n] + dBn;
    float da  = __expf(dv * An);
    aP *= da;
    h = __builtin_fmaf(da, h, dv * Bn * uv);
  }
  const size_t chain = (((size_t)dir * BB + b) * DI + d) * NS + n;
  carryA[chain * NCH + chunk] = aP;
  carryB[chain * NCH + chunk] = h;
}

__global__ void scan_stitch(const float* __restrict__ carryA,
                            const float* __restrict__ carryB,
                            float* __restrict__ h0) {
  const int chain = blockIdx.x * blockDim.x + threadIdx.x;
  if (chain >= DIRS * BB * DI * NS) return;
  const float* ca = carryA + (size_t)chain * NCH;
  const float* cb = carryB + (size_t)chain * NCH;
  float*       ho = h0     + (size_t)chain * NCH;
  float h = 0.f;
#pragma unroll
  for (int c = 0; c < NCH; ++c) {
    ho[c] = h;
    h = __builtin_fmaf(ca[c], h, cb[c]);
  }
}

__global__ void scan_pass2(const float* __restrict__ delta,
                           const float* __restrict__ xz,
                           const float* __restrict__ xdbl,
                           const float* __restrict__ A_log,
                           const float* __restrict__ dirB,
                           const float* __restrict__ Dp,
                           const int*   __restrict__ ord,
                           const float* __restrict__ h0,
                           float* __restrict__ y) {
  const int wave = blockIdx.x * (blockDim.x >> 5) + (threadIdx.x >> 5);
  const int lane = threadIdx.x & 31;
  int dir, b, d, n, chunk;
  scan_decode(wave, lane, dir, b, d, n, chunk);
  const float An  = -__expf(A_log[d * NS + n]);
  const float dBn = dirB[dir * NS + n];
  const float Dd  = Dp[d];
  const int* o = ord + dir * LL;
  const size_t chain = (((size_t)dir * BB + b) * DI + d) * NS + n;
  float h = h0[chain * NCH + chunk];
  const int t0 = chunk * CHUNK;
  for (int t = t0; t < t0 + CHUNK; ++t) {
    int   idx = o[t];
    float dv  = delta[((size_t)(b * LL + t)) * DI + d];
    float uv  = xz[((size_t)(b * LL + idx)) * (2 * DI) + d];
    float Bn  = xdbl[((size_t)(b * LL + t)) * XDBL_LD + DTR + n] + dBn;
    float da  = __expf(dv * An);
    h = __builtin_fmaf(da, h, dv * Bn * uv);
    float Cn  = xdbl[((size_t)(b * LL + t)) * XDBL_LD + DTR + NS + n];
    float contrib = Cn * h;
    contrib += __shfl_xor(contrib, 1, 16);
    contrib += __shfl_xor(contrib, 2, 16);
    contrib += __shfl_xor(contrib, 4, 16);
    contrib += __shfl_xor(contrib, 8, 16);
    if (n == 0)  // each (dir,b,idx,d) written exactly once -> plain store
      y[(((size_t)dir * ML) + (size_t)(b * LL + idx)) * DI + d] = contrib + Dd * uv;
  }
}

// ---------------- LayerNorm(y)/LayerNorm(z) softmax channel mix ----------
__global__ void ln_mix(const float* __restrict__ y,       // 4 direction buffers
                       const float* __restrict__ xz,      // z = cols 256..511
                       const float* __restrict__ lnw,
                       const float* __restrict__ lnb,
                       const float* __restrict__ bc,
                       unsigned short* __restrict__ yc) { // bf16 for out GEMM
  __shared__ float s[4][DI];
  const int m = blockIdx.x, d = threadIdx.x;
  float yv = 0.f;
#pragma unroll
  for (int dir = 0; dir < DIRS; ++dir)
    yv += y[(((size_t)dir * ML) + m) * DI + d];
  float zv = xz[(size_t)m * (2 * DI) + DI + d];
  s[0][d] = yv; s[1][d] = yv * yv; s[2][d] = zv; s[3][d] = zv * zv;
  __syncthreads();
  for (int off = DI / 2; off > 0; off >>= 1) {
    if (d < off) {
      s[0][d] += s[0][d + off]; s[1][d] += s[1][d + off];
      s[2][d] += s[2][d + off]; s[3][d] += s[3][d + off];
    }
    __syncthreads();
  }
  float my = s[0][0] * (1.f / DI), vy = s[1][0] * (1.f / DI) - my * my;
  float mz = s[2][0] * (1.f / DI), vz = s[3][0] * (1.f / DI) - mz * mz;
  float lny = (yv - my) * rsqrtf(vy + 1e-5f) * lnw[d] + lnb[d];
  float lnz = (zv - mz) * rsqrtf(vz + 1e-5f) * lnw[d] + lnb[d];
  float b0 = bc[d], b1 = bc[DI + d];
  float mx = fmaxf(b0, b1);
  float e0 = __expf(b0 - mx), e1 = __expf(b1 - mx);
  float c0 = e0 / (e0 + e1);
  yc[(size_t)m * DI + d] = f2bf(c0 * lny + (1.f - c0) * lnz);
}

// ------------------------------------------------------------------------
extern "C" void kernel_launch(void* const* d_in, const int* in_sizes, int n_in,
                              void* d_out, int out_size, void* d_ws, size_t ws_size,
                              hipStream_t stream) {
  const float* x_norm    = (const float*)d_in[0];
  const float* in_proj_w = (const float*)d_in[1];
  const float* x_proj_w  = (const float*)d_in[2];
  const float* dt_proj_w = (const float*)d_in[3];
  const float* dt_proj_b = (const float*)d_in[4];
  const float* A_log     = (const float*)d_in[5];
  const float* Dp        = (const float*)d_in[6];
  const float* dirB      = (const float*)d_in[7];
  const float* bc        = (const float*)d_in[8];
  const float* lnw       = (const float*)d_in[9];
  const float* lnb       = (const float*)d_in[10];
  const float* out_proj_w= (const float*)d_in[11];
  float* out = (float*)d_out;

  // workspace carve-up (256B aligned)
  char* ws = (char*)d_ws;
  size_t off = 0;
  auto carve = [&](size_t bytes) { char* p = ws + off; off = (off + bytes + 255) & ~(size_t)255; return p; };
  int*            ord     = (int*)            carve((size_t)DIRS * LL * 4);
  unsigned short* xnorm_bf= (unsigned short*) carve((size_t)ML * DM * 2);
  unsigned short* w_in_bf = (unsigned short*) carve((size_t)2 * DI * DM * 2);
  unsigned short* w_xp_bf = (unsigned short*) carve((size_t)XDBL_LD * DI * 2);
  unsigned short* w_out_bf= (unsigned short*) carve((size_t)DM * DI * 2);
  unsigned short* u_bf    = (unsigned short*) carve((size_t)ML * DI * 2);
  unsigned short* yc_bf   = (unsigned short*) carve((size_t)ML * DI * 2);
  float* xz     = (float*)carve((size_t)ML * 2 * DI * 4);        // u | z
  float* xdbl   = (float*)carve((size_t)ML * XDBL_LD * 4);       // dt | B | C
  float* delta  = (float*)carve((size_t)ML * DI * 4);
  float* carryA = (float*)carve((size_t)DIRS * BB * DI * NS * NCH * 4);
  float* carryB = (float*)carve((size_t)DIRS * BB * DI * NS * NCH * 4);
  float* h0     = (float*)carve((size_t)DIRS * BB * DI * NS * NCH * 4);
  float* ybuf   = (float*)carve((size_t)DIRS * ML * DI * 4);
  (void)ws_size; (void)n_in; (void)in_sizes; (void)out_size;

  prep_orders<<<1, 256, 0, stream>>>(ord);
  f32_to_bf16<<<(ML * DM + 255) / 256, 256, 0, stream>>>(x_norm, xnorm_bf, ML * DM);
  f32_to_bf16<<<(2 * DI * DM + 255) / 256, 256, 0, stream>>>(in_proj_w, w_in_bf, 2 * DI * DM);
  f32_to_bf16<<<(DM * DI + 255) / 256, 256, 0, stream>>>(out_proj_w, w_out_bf, DM * DI);
  pad_xproj_bf<<<(XDBL_LD * DI + 255) / 256, 256, 0, stream>>>(x_proj_w, w_xp_bf);

  // in_proj: (4608x128)·(512x128)^T -> xz (4608x512), NB=4 -> 2304 waves
  wmma_gemm_bf<4><<<(ML / 16) * (2 * DI / 64) / 4, 128, 0, stream>>>(
      xnorm_bf, DM, w_in_bf, DM, xz, 2 * DI, ML, 2 * DI, DM);
  extract_u_bf<<<(ML * DI + 255) / 256, 256, 0, stream>>>(xz, u_bf);

  // x_proj: (4608x256)·(48x256)^T -> x_dbl (4608x48), NB=3 -> 288 waves
  wmma_gemm_bf<3><<<(ML / 16) * (XDBL_LD / 48) / 4, 128, 0, stream>>>(
      u_bf, DI, w_xp_bf, DI, xdbl, XDBL_LD, ML, XDBL_LD, DI);

  compute_delta<<<ML, DI, 0, stream>>>(xdbl, dt_proj_w, dt_proj_b, delta);

  const int scan_blocks = (DIRS * BB * (DI / 2) * NCH) / 8;  // 8 waves/block
  scan_pass1<<<scan_blocks, 256, 0, stream>>>(delta, xz, xdbl, A_log, dirB, ord,
                                              carryA, carryB);
  scan_stitch<<<(DIRS * BB * DI * NS + 255) / 256, 256, 0, stream>>>(carryA, carryB, h0);
  scan_pass2<<<scan_blocks, 256, 0, stream>>>(delta, xz, xdbl, A_log, dirB, Dp, ord,
                                              h0, ybuf);

  ln_mix<<<ML, DI, 0, stream>>>(ybuf, xz, lnw, lnb, bc, yc_bf);

  // out_proj: (4608x256)·(128x256)^T -> out (4608x128), NB=4 -> 576 waves
  wmma_gemm_bf<4><<<(ML / 16) * (DM / 64) / 4, 128, 0, stream>>>(
      yc_bf, DI, w_out_bf, DI, out, DM, ML, DM, DI);
}
